// ScaledDotProductAttention_5858335392368
// MI455X (gfx1250) — compile-verified, matrix operand order checked by reading, not run on hardware
//
#include <hip/hip_runtime.h>

typedef __attribute__((ext_vector_type(16))) _Float16 v16h;
typedef __attribute__((ext_vector_type(8)))  float    v8f;

#define BS_  8
#define SEQ_ 2048
#define D_   128

constexpr int KV_TILE  = 64;     // keys per block step
constexpr int K_PITCH  = 136;    // halfs per K row in LDS (128 + 8 pad)
constexpr int VT_PITCH = 72;     // halfs per Vt row in LDS (64 + 8 pad)
#define MASK_VALUE (-1000000.0f)

__global__ __launch_bounds__(256)
void fa_fwd_kernel(const float* __restrict__ Q, const float* __restrict__ K,
                   const float* __restrict__ V, const int* __restrict__ VL,
                   float* __restrict__ Out) {
  const int tid  = threadIdx.x;
  const int lane = tid & 31;
  const int wave = tid >> 5;
  const int b    = blockIdx.x >> 4;             // 16 q-blocks of 128 rows per batch
  const int qrow0 = ((blockIdx.x & 15) * 128) + wave * 16;
  const int valid = VL[b];                      // uniform per block

  const int m  = lane & 15;                     // query column (N) / key-row selector
  const int hi = lane >> 4;                     // half-wave selector

  __shared__ _Float16 sK [KV_TILE * K_PITCH];   // K tile, row-major [key][d]
  __shared__ _Float16 sVt[D_ * VT_PITCH];       // V tile, transposed [d][key]

  // ---- Q^T B-fragments: 4 chunks of K=32 over D=128, 1/sqrt(d) folded in
  const float scale = 0.08838834764831845f;     // 1/sqrt(128)
  v16h qf[4];
  {
    const float* qrow = Q + (size_t)(b * SEQ_ + qrow0 + m) * D_;
    #pragma unroll
    for (int kc = 0; kc < 4; ++kc) {
      const float* p0 = qrow + kc * 32 + hi * 8;
      #pragma unroll
      for (int j = 0; j < 8; ++j) {
        qf[kc][j]     = (_Float16)(p0[j]      * scale);
        qf[kc][8 + j] = (_Float16)(p0[16 + j] * scale);
      }
    }
  }

  v8f o[8];
  #pragma unroll
  for (int dt = 0; dt < 8; ++dt) o[dt] = v8f{};
  float mi = -3.0e38f;
  float li = 0.0f;

  const float* Kb = K + (size_t)b * SEQ_ * D_;
  const float* Vb = V + (size_t)b * SEQ_ * D_;

  // Tiles with kv0 >= valid contribute exactly zero (f32 exp underflow),
  // identically to the reference softmax -> loop only while kv0 < valid.
  for (int kv0 = 0; kv0 < valid; kv0 += KV_TILE) {
    __syncthreads();
    // -------- cooperative staging: K tile (f16 row-major), V tile (f16 transposed)
    {
      const float4* ksrc = (const float4*)(Kb + (size_t)kv0 * D_);
      #pragma unroll
      for (int c = 0; c < 8; ++c) {
        int idx4 = c * 256 + tid;               // 2048 float4 = 64x128 floats
        int row  = idx4 >> 5;                   // 32 float4 per key row
        int col  = (idx4 & 31) * 4;
        float4 v = ksrc[idx4];
        _Float16* dst = sK + row * K_PITCH + col;
        dst[0] = (_Float16)v.x; dst[1] = (_Float16)v.y;
        dst[2] = (_Float16)v.z; dst[3] = (_Float16)v.w;
      }
      const float* vsrc = Vb + (size_t)kv0 * D_;
      #pragma unroll
      for (int c = 0; c < 32; ++c) {
        int idx = c * 256 + tid;                // 8192 = 64x128
        int row = idx >> 7;                     // key
        int col = idx & 127;                    // d
        sVt[col * VT_PITCH + row] = (_Float16)vsrc[idx];
      }
    }
    if (kv0 + KV_TILE < SEQ_) {                 // overlap next tile's HBM latency
      __builtin_prefetch(Kb + (size_t)(kv0 + KV_TILE) * D_ + tid * 32, 0, 1);
      __builtin_prefetch(Vb + (size_t)(kv0 + KV_TILE) * D_ + tid * 32, 0, 1);
    }
    __syncthreads();

    // -------- S^T = K_tile * Q^T : four 16-key sub-tiles (M=keys), chained over d
    v8f st[4];
    #pragma unroll
    for (int nt = 0; nt < 4; ++nt) {
      v16h kf[4];                               // A operand: K rows, regs walk d
      #pragma unroll
      for (int kc = 0; kc < 4; ++kc) {
        const _Float16* krow = sK + (nt * 16 + m) * K_PITCH + kc * 32 + hi * 8;
        #pragma unroll
        for (int j = 0; j < 8; ++j) { kf[kc][j] = krow[j]; kf[kc][8 + j] = krow[16 + j]; }
      }
      v8f acc = v8f{};
      #pragma unroll
      for (int kc = 0; kc < 4; ++kc)
        acc = __builtin_amdgcn_wmma_f32_16x16x32_f16(false, kf[kc], false, qf[kc],
                                                     (short)0, acc, false, false);
      st[nt] = acc;
    }

    // -------- padding mask: boundary tile only (uniform scalar branch).
    // Reg r of sub-tile nt holds key kv0 + nt*16 + 8*hi + r.
    if (kv0 + KV_TILE > valid) {
      const int t0 = valid - kv0 - hi * 8;
      #pragma unroll
      for (int nt = 0; nt < 4; ++nt) {
        const int t = t0 - nt * 16;
        #pragma unroll
        for (int r = 0; r < 8; ++r)
          st[nt][r] = (r >= t) ? MASK_VALUE : st[nt][r];
      }
    }

    // -------- online softmax: per-lane scalar stats (lane = one query column)
    float rowmax = st[0][0];
    #pragma unroll
    for (int nt = 0; nt < 4; ++nt)
      #pragma unroll
      for (int r = 0; r < 8; ++r)
        if (nt || r) rowmax = fmaxf(rowmax, st[nt][r]);
    rowmax = fmaxf(rowmax, __shfl_xor(rowmax, 16, 32));  // merge the two key halves
    const float mnew = fmaxf(mi, rowmax);
    const float corr = __expf(mi - mnew);
    float rs = 0.0f;
    #pragma unroll
    for (int nt = 0; nt < 4; ++nt)
      #pragma unroll
      for (int r = 0; r < 8; ++r) {
        float p = __expf(st[nt][r] - mnew);
        st[nt][r] = p;
        rs += p;
      }
    rs += __shfl_xor(rs, 16, 32);
    li = li * corr + rs;
    mi = mnew;

    // -------- P already sits in B-operand layout: pack f32->f16 in registers
    v16h pf0, pf1;   // keys 0..31 / 32..63 : b[j]=P[koff+8*hi+j][q], b[8+j]=+16
    #pragma unroll
    for (int j = 0; j < 8; ++j) {
      pf0[j]     = (_Float16)st[0][j];
      pf0[8 + j] = (_Float16)st[1][j];
      pf1[j]     = (_Float16)st[2][j];
      pf1[8 + j] = (_Float16)st[3][j];
    }

    // -------- O^T = corr*O^T + V^T * P  (A = V^T rows of d, from transposed LDS)
    #pragma unroll
    for (int dt = 0; dt < 8; ++dt) {
      const _Float16* vr = sVt + (dt * 16 + m) * VT_PITCH + hi * 8;
      v16h vf0, vf1;
      #pragma unroll
      for (int j = 0; j < 8; ++j) {
        vf0[j] = vr[j];      vf0[8 + j] = vr[16 + j];
        vf1[j] = vr[32 + j]; vf1[8 + j] = vr[48 + j];
      }
      #pragma unroll
      for (int r = 0; r < 8; ++r) o[dt][r] *= corr;
      o[dt] = __builtin_amdgcn_wmma_f32_16x16x32_f16(false, vf0, false, pf0,
                                                     (short)0, o[dt], false, false);
      o[dt] = __builtin_amdgcn_wmma_f32_16x16x32_f16(false, vf1, false, pf1,
                                                     (short)0, o[dt], false, false);
    }
  }

  // -------- epilogue: normalize, write fp32 output (O transposed: lane = query;
  // per lane r=0..7 are consecutive d -> pack into float4 stores)
  const float inv = 1.0f / li;
  float* op = Out + (size_t)(b * SEQ_ + qrow0 + m) * D_ + hi * 8;
  #pragma unroll
  for (int dt = 0; dt < 8; ++dt) {
    float4 lo  = make_float4(o[dt][0] * inv, o[dt][1] * inv,
                             o[dt][2] * inv, o[dt][3] * inv);
    float4 hi4 = make_float4(o[dt][4] * inv, o[dt][5] * inv,
                             o[dt][6] * inv, o[dt][7] * inv);
    *(float4*)(op + dt * 16)     = lo;
    *(float4*)(op + dt * 16 + 4) = hi4;
  }
}

extern "C" void kernel_launch(void* const* d_in, const int* in_sizes, int n_in,
                              void* d_out, int out_size, void* d_ws, size_t ws_size,
                              hipStream_t stream) {
  (void)in_sizes; (void)n_in; (void)out_size; (void)d_ws; (void)ws_size;
  const float* Q  = (const float*)d_in[0];
  const float* K  = (const float*)d_in[1];
  const float* V  = (const float*)d_in[2];
  const int*   VL = (const int*)d_in[3];
  float* O = (float*)d_out;

  dim3 grid(BS_ * (SEQ_ / 128));   // 128 blocks: 8 batches x 16 q-tiles of 128 rows
  dim3 block(256);                 // 8 wave32 -> one 16-row q-tile each
  fa_fwd_kernel<<<grid, block, 0, stream>>>(Q, K, V, VL, O);
}
